// InteractionGraphEncoder_21466246545470
// MI455X (gfx1250) — matrix operand order repaired and checked by reading, not compiled, not run on hardware
//
#include <hip/hip_runtime.h>
#include <hip/hip_fp16.h>

// ---------------------------------------------------------------------------
// InteractionGraphEncoder for MI455X (gfx1250, wave32, WMMA)
//
// Dense GEMMs (x@W per relation, final projection) run on
// v_wmma_f32_16x16x32_f16 (f16 in, f32 accumulate). B tiles are staged in LDS
// pre-swizzled into WMMA fragment order so each wave fetches its B fragment
// with two ds_load_b128 (no per-half repacking). Edge softmax/scatter is
// bandwidth-bound and uses 3-pass segment softmax with float atomics.
// ---------------------------------------------------------------------------

typedef __attribute__((ext_vector_type(16))) _Float16 v16h;
typedef __attribute__((ext_vector_type(8)))  float    v8f;

#define NNODES 100000
#define NEDGES 500000
#define NREL   6
#define INDIM  64
#define HIDDIM 128
#define NHEADS 4
#define NGRAPH 1024

// ---------------------------------------------------------------- utilities
__global__ void k_cvt_f16(const float* __restrict__ in, _Float16* __restrict__ out, int n) {
  int i = blockIdx.x * blockDim.x + threadIdx.x;
  if (i < n) out[i] = (_Float16)in[i];
}

__global__ void k_fill(float* __restrict__ p, float v, int n) {
  int i = blockIdx.x * blockDim.x + threadIdx.x;
  if (i < n) p[i] = v;
}

__device__ __forceinline__ void atomicMaxF(float* addr, float v) {
  // IEEE ordering trick: int compare for >=0, reversed uint compare for <0.
  if (v >= 0.f) atomicMax((int*)addr, __float_as_int(v));
  else          atomicMin((unsigned int*)addr, (unsigned int)__float_as_int(v));
}

__device__ __forceinline__ float leaky02(float x) { return x > 0.f ? x : 0.2f * x; }

// ------------------------------------------------------------- WMMA GEMM
// D[r] = A @ B[r] (+bias). A:[M,K] f16 row-major. B:[K,128] f16 row-major per
// relation (stride K*128). D:[M,128] f32 per relation (stride M*128).
// Block = 128 threads = 4 waves; wave w owns 16-row tile (blockIdx.x*4+w).
//
// B tile (K x 16) is staged in LDS in WMMA *fragment order*:
//   Bs[slab][lane][t],  slab = k/32, lane = j + (((k/8)%2) ? 16 : 0),
//   t = k%8 + 8*((k%32)>=16)   (B 32x16 f16 layout, ISA 7.12.2)
// so each wave reads its fragment with two 16B LDS loads per K-slab.
__global__ __launch_bounds__(128) void k_wmma_gemm(
    const _Float16* __restrict__ A, const _Float16* __restrict__ B,
    float* __restrict__ D, const float* __restrict__ bias, int M, int K)
{
  __shared__ _Float16 Bs[4 * 32 * 16];   // up to 4 K-slabs x 32 lanes x 16 halves
  const int tid  = threadIdx.x;
  const int lane = tid & 31;
  const int nt   = blockIdx.y;
  const int r    = blockIdx.z;
  const _Float16* Br = B + (size_t)r * K * HIDDIM;
  float*          Dr = D + (size_t)r * M * HIDDIM;

  // stage B tile, swizzled to fragment order
  for (int i = tid; i < K * 16; i += 128) {
    int k = i >> 4, j = i & 15;
    int slab = k >> 5;
    int l    = j + (((k >> 3) & 1) ? 16 : 0);
    int t    = (k & 7) | ((k >> 1) & 8);
    Bs[(slab * 32 + l) * 16 + t] = Br[(size_t)k * HIDDIM + nt * 16 + j];
  }
  __syncthreads();

  // clamp last partial block's waves onto a valid tile (benign duplicate write)
  int mt = blockIdx.x * 4 + (tid >> 5);
  const int mtiles = M / 16;
  if (mt >= mtiles) mt = mtiles - 1;

  // A fragment addressing (16-bit A 16x32 layout, ISA 7.12.2)
  const int arow = mt * 16 + (lane & 15);
  const int ak   = (lane < 16) ? 0 : 8;    // K start within 32-wide slab

  v8f c;
#pragma unroll
  for (int i = 0; i < 8; ++i) c[i] = 0.f;

  for (int kk = 0, slab = 0; kk < K; kk += 32, ++slab) {
    union { v16h v; uint4 q[2]; } a;
    const uint4* ap = reinterpret_cast<const uint4*>(A + (size_t)arow * K + kk + ak);
    a.q[0] = ap[0];   // K = kk+ak .. +7
    a.q[1] = ap[2];   // K = kk+ak+16 .. +23
    union { v16h v; uint4 q[2]; } b;
    const uint4* bp = reinterpret_cast<const uint4*>(Bs + (slab * 32 + lane) * 16);
    b.q[0] = bp[0];
    b.q[1] = bp[1];
    c = __builtin_amdgcn_wmma_f32_16x16x32_f16(
        /*neg_a=*/false, a.v, /*neg_b=*/false, b.v,
        /*c_mod=*/(short)0, c, /*reuse_a=*/false, /*reuse_b=*/false);
  }

  const int drow = mt * 16 + ((lane < 16) ? 0 : 8);
  const int dcol = nt * 16 + (lane & 15);
  const float bv = bias ? bias[dcol] : 0.f;
#pragma unroll
  for (int i = 0; i < 8; ++i)
    Dr[(size_t)(drow + i) * HIDDIM + dcol] = c[i] + bv;
}

// ---------------------------------------------------- attention logits
// als[r][n][h] = dot(H[r][n][h*32:...], a_src[r][h]); same for ald.
__global__ void k_attn_logits(const float* __restrict__ H,
                              const float* __restrict__ a_s,
                              const float* __restrict__ a_d,
                              float* __restrict__ als, float* __restrict__ ald)
{
  int idx = blockIdx.x * blockDim.x + threadIdx.x;  // [R*N*H]
  if (idx >= NREL * NNODES * NHEADS) return;
  int h = idx & 3;
  int n = (idx >> 2) % NNODES;
  int r = (idx >> 2) / NNODES;
  const float* hp  = H   + ((size_t)r * NNODES + n) * HIDDIM + h * 32;
  const float* asp = a_s + (r * NHEADS + h) * 32;
  const float* adp = a_d + (r * NHEADS + h) * 32;
  float ss = 0.f, sd = 0.f;
#pragma unroll 8
  for (int ccc = 0; ccc < 32; ++ccc) { ss += hp[ccc] * asp[ccc]; sd += hp[ccc] * adp[ccc]; }
  als[idx] = ss;
  ald[idx] = sd;
}

// ------------------------------------------------------- edge passes
__global__ void k_edge_max(const int* __restrict__ ei,
                           const float* __restrict__ als, const float* __restrict__ ald,
                           float* __restrict__ m)
{
  int idx = blockIdx.x * blockDim.x + threadIdx.x;  // [R*E*H]
  if (idx >= NREL * NEDGES * NHEADS) return;
  int h = idx & 3;
  int e = (idx >> 2) % NEDGES;
  int r = (idx >> 2) / NEDGES;
  int src = ei[(size_t)(r * 2)     * NEDGES + e];
  int dst = ei[(size_t)(r * 2 + 1) * NEDGES + e];
  float ev = leaky02(als[((size_t)r * NNODES + src) * 4 + h] +
                     ald[((size_t)r * NNODES + dst) * 4 + h]);
  atomicMaxF(&m[((size_t)r * NNODES + dst) * 4 + h], ev);
}

__global__ void k_edge_exp(const int* __restrict__ ei,
                           const float* __restrict__ als, const float* __restrict__ ald,
                           const float* __restrict__ m, float* __restrict__ den,
                           float* __restrict__ exbuf)
{
  int idx = blockIdx.x * blockDim.x + threadIdx.x;  // [R*E*H]
  if (idx >= NREL * NEDGES * NHEADS) return;
  int h = idx & 3;
  int e = (idx >> 2) % NEDGES;
  int r = (idx >> 2) / NEDGES;
  int src = ei[(size_t)(r * 2)     * NEDGES + e];
  int dst = ei[(size_t)(r * 2 + 1) * NEDGES + e];
  size_t di = ((size_t)r * NNODES + dst) * 4 + h;
  float ev = leaky02(als[((size_t)r * NNODES + src) * 4 + h] +
                     ald[((size_t)r * NNODES + dst) * 4 + h]);
  float ex = __expf(ev - m[di]);
  exbuf[((size_t)r * NEDGES + e) * 4 + h] = ex;
  atomicAdd(&den[di], ex);
}

__global__ void k_edge_scatter(const int* __restrict__ ei,
                               const float* __restrict__ H,
                               const float* __restrict__ exbuf,
                               const float* __restrict__ den,
                               float* __restrict__ acc)
{
  int idx = blockIdx.x * blockDim.x + threadIdx.x;  // [R*E*32] (4 ch each)
  if (idx >= NREL * NEDGES * 32) return;
  int c4 = idx & 31;
  int e  = (idx >> 5) % NEDGES;
  int r  = (idx >> 5) / NEDGES;
  int h  = c4 >> 3;
  int ch = c4 * 4;
  int src = ei[(size_t)(r * 2)     * NEDGES + e];
  int dst = ei[(size_t)(r * 2 + 1) * NEDGES + e];
  float ex    = exbuf[((size_t)r * NEDGES + e) * 4 + h];
  float alpha = ex / (den[((size_t)r * NNODES + dst) * 4 + h] + 1e-16f);
  float4 hv = *reinterpret_cast<const float4*>(H + ((size_t)r * NNODES + src) * HIDDIM + ch);
  float* op = acc + (size_t)dst * HIDDIM + ch;
  atomicAdd(op + 0, hv.x * alpha);
  atomicAdd(op + 1, hv.y * alpha);
  atomicAdd(op + 2, hv.z * alpha);
  atomicAdd(op + 3, hv.w * alpha);
}

// ------------------------------------------- softsign + layernorm + f16 copy
__global__ __launch_bounds__(128) void k_node_post(
    const float* __restrict__ acc, const float* __restrict__ b6,   // [6,128]
    const float* __restrict__ g, const float* __restrict__ be,
    float* __restrict__ hcur, _Float16* __restrict__ hf16)
{
  int node = blockIdx.x * 4 + (threadIdx.x >> 5);
  if (node >= NNODES) return;
  int lane = threadIdx.x & 31;
  float v[4], s = 0.f;
#pragma unroll
  for (int j = 0; j < 4; ++j) {
    int ccc = j * 32 + lane;
    float bs = b6[ccc] + b6[128 + ccc] + b6[256 + ccc] + b6[384 + ccc] + b6[512 + ccc] + b6[640 + ccc];
    float t = acc[(size_t)node * HIDDIM + ccc] + bs;
    t = t / (1.f + fabsf(t));          // softsign
    v[j] = t; s += t;
  }
#pragma unroll
  for (int o = 16; o > 0; o >>= 1) s += __shfl_xor(s, o, 32);
  float mu = s * (1.f / 128.f);
  float var = 0.f;
#pragma unroll
  for (int j = 0; j < 4; ++j) { float d = v[j] - mu; var += d * d; }
#pragma unroll
  for (int o = 16; o > 0; o >>= 1) var += __shfl_xor(var, o, 32);
  float rs = rsqrtf(var * (1.f / 128.f) + 1e-5f);
#pragma unroll
  for (int j = 0; j < 4; ++j) {
    int ccc = j * 32 + lane;
    float y = (v[j] - mu) * rs * g[ccc] + be[ccc];
    hcur[(size_t)node * HIDDIM + ccc] = y;
    hf16[(size_t)node * HIDDIM + ccc] = (_Float16)y;
  }
}

// --------------------------------------------------------- pooling
__global__ __launch_bounds__(128) void k_scores(const float* __restrict__ hcur,
                                                const float* __restrict__ q,
                                                float* __restrict__ scores)
{
  int node = blockIdx.x * 4 + (threadIdx.x >> 5);
  if (node >= NNODES) return;
  int lane = threadIdx.x & 31;
  float s = 0.f;
#pragma unroll
  for (int j = 0; j < 4; ++j) {
    int ccc = j * 32 + lane;
    s += hcur[(size_t)node * HIDDIM + ccc] * q[ccc];
  }
#pragma unroll
  for (int o = 16; o > 0; o >>= 1) s += __shfl_xor(s, o, 32);
  if (lane == 0) scores[node] = s;
}

__global__ void k_pool_max(const float* __restrict__ scores, const int* __restrict__ batch,
                           float* __restrict__ gm)
{
  int n = blockIdx.x * blockDim.x + threadIdx.x;
  if (n < NNODES) atomicMaxF(&gm[batch[n]], scores[n]);
}

__global__ void k_pool_exp(const float* __restrict__ scores, const int* __restrict__ batch,
                           const float* __restrict__ gm, float* __restrict__ gden,
                           float* __restrict__ exs)
{
  int n = blockIdx.x * blockDim.x + threadIdx.x;
  if (n >= NNODES) return;
  int g = batch[n];
  float ex = __expf(scores[n] - gm[g]);
  exs[n] = ex;
  atomicAdd(&gden[g], ex);
}

__global__ void k_pool_scatter(const float* __restrict__ hcur, const float* __restrict__ exs,
                               const float* __restrict__ gden, const int* __restrict__ batch,
                               float* __restrict__ pooled)
{
  int idx = blockIdx.x * blockDim.x + threadIdx.x;  // [N*32] (4 ch each)
  if (idx >= NNODES * 32) return;
  int c4 = idx & 31;
  int n  = idx >> 5;
  int g  = batch[n];
  float w = exs[n] / (gden[g] + 1e-16f);
  int ch = c4 * 4;
  float4 hv = *reinterpret_cast<const float4*>(hcur + (size_t)n * HIDDIM + ch);
  float* op = pooled + (size_t)g * HIDDIM + ch;
  atomicAdd(op + 0, hv.x * w);
  atomicAdd(op + 1, hv.y * w);
  atomicAdd(op + 2, hv.z * w);
  atomicAdd(op + 3, hv.w * w);
}

// ===========================================================================
extern "C" void kernel_launch(void* const* d_in, const int* in_sizes, int n_in,
                              void* d_out, int out_size, void* d_ws, size_t ws_size,
                              hipStream_t stream) {
  (void)in_sizes; (void)n_in; (void)out_size; (void)ws_size;
  const float* x    = (const float*)d_in[0];
  const int*   ei   = (const int*)  d_in[1];
  const int*   batch= (const int*)  d_in[2];
  const float* W1   = (const float*)d_in[3];
  const float* as1  = (const float*)d_in[4];
  const float* ad1  = (const float*)d_in[5];
  const float* b1   = (const float*)d_in[6];
  const float* W2   = (const float*)d_in[7];
  const float* as2  = (const float*)d_in[8];
  const float* ad2  = (const float*)d_in[9];
  const float* b2   = (const float*)d_in[10];
  const float* g1   = (const float*)d_in[11];
  const float* be1  = (const float*)d_in[12];
  const float* g2   = (const float*)d_in[13];
  const float* be2  = (const float*)d_in[14];
  const float* q    = (const float*)d_in[15];
  const float* Wp   = (const float*)d_in[16];
  const float* bp   = (const float*)d_in[17];
  float* out = (float*)d_out;

  // ---- workspace carve-out
  char* ws = (char*)d_ws;
  size_t off = 0;
  auto alloc = [&](size_t bytes) -> void* {
    void* p = ws + off;
    off = (off + bytes + 255) & ~(size_t)255;
    return p;
  };
  _Float16* xf16    = (_Float16*)alloc((size_t)NNODES * INDIM * 2);
  _Float16* hf16    = (_Float16*)alloc((size_t)NNODES * HIDDIM * 2);
  _Float16* W1h     = (_Float16*)alloc((size_t)NREL * INDIM * HIDDIM * 2);
  _Float16* W2h     = (_Float16*)alloc((size_t)NREL * HIDDIM * HIDDIM * 2);
  _Float16* Wph     = (_Float16*)alloc((size_t)HIDDIM * HIDDIM * 2);
  float* H      = (float*)alloc((size_t)NREL * NNODES * HIDDIM * 4);
  float* als    = (float*)alloc((size_t)NREL * NNODES * NHEADS * 4);
  float* ald    = (float*)alloc((size_t)NREL * NNODES * NHEADS * 4);
  float* mbuf   = (float*)alloc((size_t)NREL * NNODES * NHEADS * 4);
  float* den    = (float*)alloc((size_t)NREL * NNODES * NHEADS * 4);
  float* exbuf  = (float*)alloc((size_t)NREL * NEDGES * NHEADS * 4);
  float* acc    = (float*)alloc((size_t)NNODES * HIDDIM * 4);
  float* hcur   = (float*)alloc((size_t)NNODES * HIDDIM * 4);
  float* scores = (float*)alloc((size_t)NNODES * 4);
  float* exs    = (float*)alloc((size_t)NNODES * 4);
  float* gm     = (float*)alloc((size_t)NGRAPH * 4);
  float* gden   = (float*)alloc((size_t)NGRAPH * 4);
  float* pooled = (float*)alloc((size_t)NGRAPH * HIDDIM * 4);
  _Float16* pooledh = (_Float16*)alloc((size_t)NGRAPH * HIDDIM * 2);

  const int TPB = 256;
  auto cdiv = [](long long a, long long b) { return (int)((a + b - 1) / b); };

  // ---- f16 mirrors
  k_cvt_f16<<<cdiv((long long)NNODES * INDIM, TPB), TPB, 0, stream>>>(x,  xf16, NNODES * INDIM);
  k_cvt_f16<<<cdiv((long long)NREL * INDIM * HIDDIM, TPB), TPB, 0, stream>>>(W1, W1h, NREL * INDIM * HIDDIM);
  k_cvt_f16<<<cdiv((long long)NREL * HIDDIM * HIDDIM, TPB), TPB, 0, stream>>>(W2, W2h, NREL * HIDDIM * HIDDIM);
  k_cvt_f16<<<cdiv((long long)HIDDIM * HIDDIM, TPB), TPB, 0, stream>>>(Wp, Wph, HIDDIM * HIDDIM);

  auto run_layer = [&](const _Float16* Ain, int K, const _Float16* Wh,
                       const float* a_s, const float* a_d, const float* b,
                       const float* gg, const float* bee) {
    dim3 grid(cdiv(NNODES / 16, 4), HIDDIM / 16, NREL);
    k_wmma_gemm<<<grid, 128, 0, stream>>>(Ain, Wh, H, nullptr, NNODES, K);
    int nlog = NREL * NNODES * NHEADS;
    k_attn_logits<<<cdiv(nlog, TPB), TPB, 0, stream>>>(H, a_s, a_d, als, ald);
    k_fill<<<cdiv(nlog, TPB), TPB, 0, stream>>>(mbuf, -3.4e38f, nlog);
    k_fill<<<cdiv(nlog, TPB), TPB, 0, stream>>>(den, 0.f, nlog);
    k_fill<<<cdiv((long long)NNODES * HIDDIM, TPB), TPB, 0, stream>>>(acc, 0.f, NNODES * HIDDIM);
    int ne = NREL * NEDGES * NHEADS;
    k_edge_max<<<cdiv(ne, TPB), TPB, 0, stream>>>(ei, als, ald, mbuf);
    k_edge_exp<<<cdiv(ne, TPB), TPB, 0, stream>>>(ei, als, ald, mbuf, den, exbuf);
    long long ns = (long long)NREL * NEDGES * 32;
    k_edge_scatter<<<cdiv(ns, TPB), TPB, 0, stream>>>(ei, H, exbuf, den, acc);
    k_node_post<<<cdiv(NNODES, 4), 128, 0, stream>>>(acc, b, gg, bee, hcur, hf16);
  };

  // ---- layer 1 (K=64), layer 2 (K=128)
  run_layer(xf16, INDIM,  W1h, as1, ad1, b1, g1, be1);
  run_layer(hf16, HIDDIM, W2h, as2, ad2, b2, g2, be2);

  // ---- attention pooling over graphs
  k_scores<<<cdiv(NNODES, 4), 128, 0, stream>>>(hcur, q, scores);
  k_fill<<<cdiv(NGRAPH, TPB), TPB, 0, stream>>>(gm, -3.4e38f, NGRAPH);
  k_fill<<<cdiv(NGRAPH, TPB), TPB, 0, stream>>>(gden, 0.f, NGRAPH);
  k_pool_max<<<cdiv(NNODES, TPB), TPB, 0, stream>>>(scores, batch, gm);
  k_pool_exp<<<cdiv(NNODES, TPB), TPB, 0, stream>>>(scores, batch, gm, gden, exs);
  k_fill<<<cdiv((long long)NGRAPH * HIDDIM, TPB), TPB, 0, stream>>>(pooled, 0.f, NGRAPH * HIDDIM);
  k_pool_scatter<<<cdiv((long long)NNODES * 32, TPB), TPB, 0, stream>>>(hcur, exs, gden, batch, pooled);

  // ---- final projection: out = pooled @ Wp + bp  (WMMA)
  k_cvt_f16<<<cdiv((long long)NGRAPH * HIDDIM, TPB), TPB, 0, stream>>>(pooled, pooledh, NGRAPH * HIDDIM);
  dim3 fgrid(cdiv(NGRAPH / 16, 4), HIDDIM / 16, 1);
  k_wmma_gemm<<<fgrid, 128, 0, stream>>>(pooledh, Wph, out, bp, NGRAPH, HIDDIM);
}
